// SpikingRNNBase_25993142075988
// MI455X (gfx1250) — matrix operand order repaired and checked by reading, not compiled, not run on hardware
//
#include <hip/hip_runtime.h>

// ---------------------------------------------------------------------------
// SpikingRNN on MI455X (gfx1250):
//   Phase 1: prep — Wcat = bf16([W_ih | W_hh]) [512 x 1024], bsum = b_ih+b_hh
//   Phase 2: gemm — x[32768,512] = bf16(A) @ Wcat^T + bsum
//            WG = 16 M-rows, A-tile staged once in LDS (bf16), 8 waves split N.
//            v_wmma_f32_16x16x32_bf16, A frags via ds_load_b128.
//   Phase 3: scan — sequential LIF update over T per (b,h) lane
// ---------------------------------------------------------------------------

typedef __bf16 bf16_t;
typedef bf16_t bf16x16 __attribute__((ext_vector_type(16)));
typedef float  f32x8   __attribute__((ext_vector_type(8)));

union BFrag {
    bf16x16        v;
    unsigned short s[16];
    uint4          u[2];
};

__device__ __forceinline__ unsigned short to_bf16_bits(float f) {
    bf16_t h = (bf16_t)f;                       // native v_cvt_*_bf16_f32
    return __builtin_bit_cast(unsigned short, h);
}

// ---------------------------------------------------------------------------
// Phase 1: weight concat + bf16 convert, bias sum.  512*1024 threads.
// ---------------------------------------------------------------------------
__global__ void snn_prep_kernel(const float* __restrict__ W_ih,
                                const float* __restrict__ b_ih,
                                const float* __restrict__ W_hh,
                                const float* __restrict__ b_hh,
                                unsigned short* __restrict__ wcat,
                                float* __restrict__ bsum) {
    const int idx = blockIdx.x * blockDim.x + threadIdx.x;   // [0, 512*1024)
    const int h = idx >> 10;
    const int k = idx & 1023;
    const float w = (k < 512) ? W_ih[h * 512 + k] : W_hh[h * 512 + (k - 512)];
    wcat[idx] = to_bf16_bits(w);
    if (idx < 512) bsum[idx] = b_ih[idx] + b_hh[idx];
}

// ---------------------------------------------------------------------------
// Phase 2: WMMA GEMM.  M=32768 (T*B), N=512 (H), K=1024 ([input|hidden]).
// Block: 256 threads = 8 waves.  blockIdx.x -> 16-row M tile.
// Wave w -> columns [w*64, w*64+64) as 4 WMMA C tiles.
// A tile (16 x 1024) converted to bf16 in LDS once; A is read from HBM once.
// ---------------------------------------------------------------------------
#define A_PITCH 1032   // 1024 + 8 bf16 pad: 2064 B row = 516 dwords = 4 mod 64 banks

__global__ void __launch_bounds__(256)
snn_gemm_kernel(const float* __restrict__ input,
                const float* __restrict__ hidden,
                const unsigned short* __restrict__ wcat,
                const float* __restrict__ bsum,
                float* __restrict__ xbuf) {
    __shared__ unsigned short lAs[16 * A_PITCH];   // 33 KB

    const int tid  = threadIdx.x;
    const int lane = tid & 31;
    const int wave = tid >> 5;                 // wave32 on CDNA5
    const int m0   = blockIdx.x * 16;
    const int n0   = wave * 64;

    const int mlo = lane & 15;
    const int hi8 = (lane >> 4) * 8;           // 0 for lanes 0-15, 8 for 16-31

    // ---- Cooperative fill: convert A tile [16 rows x 1024 k] fp32 -> bf16 LDS
    // One float4 per thread per row; k = tid*4 never straddles the 512 boundary.
    {
        const int k = tid * 4;                 // [0, 1024)
        const int ksrc = (k < 512) ? k : (k - 512);
        const float* srcbase = (k < 512) ? input : hidden;
        #pragma unroll
        for (int r = 0; r < 16; ++r) {
            const float4 f = *(const float4*)(srcbase + (size_t)(m0 + r) * 512 + ksrc);
            union { unsigned short s[4]; uint2 u; } t;
            t.s[0] = to_bf16_bits(f.x);
            t.s[1] = to_bf16_bits(f.y);
            t.s[2] = to_bf16_bits(f.z);
            t.s[3] = to_bf16_bits(f.w);
            *(uint2*)&lAs[r * A_PITCH + k] = t.u;
        }
    }
    __syncthreads();

    f32x8 acc[4];
    #pragma unroll
    for (int j = 0; j < 4; ++j)
        #pragma unroll
        for (int r = 0; r < 8; ++r) acc[j][r] = 0.0f;

    const unsigned short* arow = &lAs[mlo * A_PITCH];

    for (int kk = 0; kk < 1024; kk += 32) {
        // ---- A fragment from LDS (16-bit A 16x32 layout, ISA 7.12.2) ----
        // lane<16: K = kk+0..7, kk+16..23 ; lane>=16: +8
        const unsigned short* ap = arow + kk + hi8;
        BFrag af;
        af.u[0] = *(const uint4*)(ap + 0);     // ds_load_b128
        af.u[1] = *(const uint4*)(ap + 16);    // ds_load_b128

        // ---- B fragments from global (L2-resident Wcat) ----
        // col = lane&15 ; lane<16: K = kk+0..7, kk+16..23 ; lane>=16: +8
        const int kw = kk + hi8;
        #pragma unroll
        for (int j = 0; j < 4; ++j) {
            const unsigned short* bp =
                wcat + (size_t)(n0 + 16 * j + mlo) * 1024 + kw;
            BFrag bf;
            bf.u[0] = *(const uint4*)(bp + 0);
            bf.u[1] = *(const uint4*)(bp + 16);
            acc[j] = __builtin_amdgcn_wmma_f32_16x16x32_bf16(
                false, af.v, false, bf.v, (short)0, acc[j], false, false);
        }
    }

    // Epilogue: bias add + store.  C layout: VGPR r -> row m0+r+hi8, col n0+16j+mlo
    #pragma unroll
    for (int j = 0; j < 4; ++j) {
        const int col = n0 + 16 * j + mlo;
        const float bj = bsum[col];
        float* outp = xbuf + (size_t)(m0 + hi8) * 512 + col;
        #pragma unroll
        for (int r = 0; r < 8; ++r)
            outp[(size_t)r * 512] = acc[j][r] + bj;
    }
}

// ---------------------------------------------------------------------------
// Phase 3: sequential LIF scan.  One thread per (b,h); idx = b*512+h.
// v += (x - v)/tau (tau=2);  spike = (v >= 1);  hard reset to 0.
// ---------------------------------------------------------------------------
__global__ void snn_scan_kernel(const float* __restrict__ xbuf,
                                float* __restrict__ spikes) {
    const int idx = blockIdx.x * blockDim.x + threadIdx.x;   // [0, 32768)
    const float* xp = xbuf + idx;
    float* op = spikes + idx;
    float v = 0.0f;
    for (int t = 0; t < 512; ++t) {
        const float x = xp[(size_t)t * 32768];
        v = v + (x - v) * 0.5f;
        const bool fire = (v >= 1.0f);
        op[(size_t)t * 32768] = fire ? 1.0f : 0.0f;
        v = fire ? 0.0f : v;
    }
}

// ---------------------------------------------------------------------------
extern "C" void kernel_launch(void* const* d_in, const int* in_sizes, int n_in,
                              void* d_out, int out_size, void* d_ws, size_t ws_size,
                              hipStream_t stream) {
    const float* input  = (const float*)d_in[0];   // [512,64,512]
    const float* hidden = (const float*)d_in[1];   // [512,64,512]
    const float* W_ih   = (const float*)d_in[2];   // [512,512]
    const float* b_ih   = (const float*)d_in[3];   // [512]
    const float* W_hh   = (const float*)d_in[4];   // [512,512]
    const float* b_hh   = (const float*)d_in[5];   // [512]
    float* spikes = (float*)d_out;                 // [512,64,512] f32

    // Workspace layout: bsum (4KB pad) | Wcat bf16 (1MB) | x f32 (64MB)
    char* ws = (char*)d_ws;
    float*          bsum = (float*)ws;
    unsigned short* wcat = (unsigned short*)(ws + 4096);
    float*          xbuf = (float*)(ws + 4096 + (1u << 20));

    snn_prep_kernel<<<2048, 256, 0, stream>>>(W_ih, b_ih, W_hh, b_hh, wcat, bsum);

    snn_gemm_kernel<<<2048, 256, 0, stream>>>(input, hidden, wcat, bsum, xbuf);

    snn_scan_kernel<<<128, 256, 0, stream>>>(xbuf, spikes);
}
